// Net_25847113187870
// MI455X (gfx1250) — matrix operand order repaired and checked by reading, not compiled, not run on hardware
//
#include <hip/hip_runtime.h>
#include <hip/hip_fp16.h>

typedef __attribute__((ext_vector_type(16))) _Float16 v16h;
typedef __attribute__((ext_vector_type(8)))  float    v8f;

union Frag16 { v16h v; _Float16 e[16]; };
union Acc8   { v8f  v; float    f[8];  };

#define WAVES 4   // 4 waves per block, each wave handles a 16-sample tile

// ---------------------------------------------------------------------------
// Kernel 1: per-sample features + fc11/fc12 via WMMA + heads.
// One wave == one tile of 16 samples. Lanes 0-15 build branch-0 (normal)
// features, lanes 16-31 build branch-1 (mirrored) features; then the whole
// wave runs both GEMM pipelines with v_wmma_f32_16x16x32_f16.
// ---------------------------------------------------------------------------
__global__ __launch_bounds__(128) void mlp_kernel(
    const float* __restrict__ x0, const float* __restrict__ x1m,
    const float* __restrict__ x2, const float* __restrict__ x3,
    const int*   __restrict__ idx,
    const float* __restrict__ w00, const float* __restrict__ b00,
    const float* __restrict__ w10, const float* __restrict__ b10,
    const float* __restrict__ w1,  const float* __restrict__ b1,
    const float* __restrict__ w2,  const float* __restrict__ b2,
    const float* __restrict__ piw, const float* __restrict__ pib,
    const float* __restrict__ epw, const float* __restrict__ epb,
    const float* __restrict__ amp,
    const float* __restrict__ oh0, const float* __restrict__ oh1,
    const float* __restrict__ oh2, const float* __restrict__ oh3,
    const float* __restrict__ oh4,
    float* __restrict__ open_out,      // == d_out + N   (N x 2)
    float* __restrict__ ep_arr,        // ws: sh_price per sample
    float* __restrict__ ind_arr,       // ws: inday per sample
    int Ntiles, int N)
{
    __shared__ _Float16 sW1[96 * 128];            // fc11_w^T, K-padded, f16
    __shared__ _Float16 sW2[128 * 64];            // fc12_w^T, f16
    __shared__ _Float16 sH [WAVES][2][16][96];    // features per wave/branch
    __shared__ _Float16 sY1[WAVES][16][128];      // fc11 output (relu), f16
    __shared__ float    sY0[WAVES][2][16][64];    // fc12 output (relu), f32

    const int t    = threadIdx.x;
    const int wv   = t >> 5;
    const int lane = t & 31;
    const int tile = blockIdx.x * WAVES + wv;
    const bool active = (tile < Ntiles);

    // ---- stage weights to LDS as f16 (cooperative, whole block) ----
    for (int i = t; i < 96 * 128; i += 128) {
        int k = i >> 7, n = i & 127;
        sW1[i] = (_Float16)((k < 75) ? w1[n * 75 + k] : 0.0f);
    }
    for (int i = t; i < 128 * 64; i += 128) {
        int k = i >> 6, n = i & 63;
        sW2[i] = (_Float16)w2[n * 128 + k];
    }
    __syncthreads();

    const int lm = lane & 15;   // sample-in-tile / n-index
    const int lh = lane >> 4;   // half select

    // ---- build the 75-feature vector (lane half picks branch) ----
    if (active) {
        const int  br  = lh;
        const int  m   = lm;
        const int  s   = tile * 16 + m;
        if (s < N) {
            const float* xs  = br ? x1m : x0;
            const float  sgn = br ? -1.0f : 1.0f;

            for (int g = 0; g < 5; ++g) {                     // conv cells
                const float* w  = (g == 0) ? w00 : w10;
                const float* bb = (g == 0) ? b00 : b10;
                const float* xin = xs + ((long)s * 6 + g) * 10;
                float xv[10];
                #pragma unroll
                for (int i = 0; i < 10; ++i) xv[i] = xin[i];
                #pragma unroll
                for (int o = 0; o < 8; ++o) {
                    float a = bb[o];
                    #pragma unroll
                    for (int i = 0; i < 10; ++i) a += xv[i] * w[o * 10 + i];
                    a = (a >= 0.0f) ? a : 0.1f * a;           // leaky
                    sH[wv][br][m][g * 8 + o] = (_Float16)a;
                }
            }
            const float* x5 = xs + ((long)s * 6 + 5) * 10;    // raw channel 5
            #pragma unroll
            for (int i = 0; i < 10; ++i) sH[wv][br][m][40 + i] = (_Float16)x5[i];

            int i0 = idx[s*5], i1 = idx[s*5+1], i2 = idx[s*5+2],
                i3 = idx[s*5+3], i4 = idx[s*5+4];
            sH[wv][br][m][50] = (_Float16)oh0[i0*2+0];
            sH[wv][br][m][51] = (_Float16)oh0[i0*2+1];
            sH[wv][br][m][52] = (_Float16)oh1[i1*2+0];
            sH[wv][br][m][53] = (_Float16)oh1[i1*2+1];
            sH[wv][br][m][54] = (_Float16)oh2[i2*2+0];
            sH[wv][br][m][55] = (_Float16)oh2[i2*2+1];
            sH[wv][br][m][56] = (_Float16)oh3[i3*3+0];
            sH[wv][br][m][57] = (_Float16)oh3[i3*3+1];
            sH[wv][br][m][58] = (_Float16)oh3[i3*3+2];
            float m4 = 0.0f;
            #pragma unroll
            for (int i = 0; i < 10; ++i) m4 += oh4[i];
            m4 *= 0.1f;
            sH[wv][br][m][59] = (_Float16)(sgn * (oh4[i4] - m4));
            for (int j = 1; j < 8;  ++j) sH[wv][br][m][60 + j - 1] = (_Float16)(sgn * x2[(long)s*8  + j]);
            for (int j = 2; j < 10; ++j) sH[wv][br][m][67 + j - 2] = (_Float16)(x3[(long)s*10 + j]);
        } else {
            for (int k = 0; k < 75; ++k) sH[wv][br][m][k] = (_Float16)0.0f;
        }
        for (int k = 75; k < 96; ++k) sH[wv][br][m][k] = (_Float16)0.0f;  // K pad
    }
    __builtin_amdgcn_wave_barrier();   // wave-internal LDS ordering (DS in-order)

    if (active) {
        for (int br = 0; br < 2; ++br) {
            // ---------------- GEMM1: [16 x 96] x [96 x 128] ----------------
            Acc8 acc[8];
            #pragma unroll
            for (int nb = 0; nb < 8; ++nb) {
                float bv = b1[nb * 16 + lm];
                #pragma unroll
                for (int v = 0; v < 8; ++v) acc[nb].f[v] = bv;
            }
            for (int kb = 0; kb < 3; ++kb) {
                Frag16 A;
                #pragma unroll
                for (int v = 0; v < 8; ++v) {
                    int k0 = kb * 32 + (v >> 2) * 16 + lh * 8 + (v & 3) * 2;
                    A.e[2*v]   = sH[wv][br][lm][k0];
                    A.e[2*v+1] = sH[wv][br][lm][k0 + 1];
                }
                const int kB = kb * 32 + lane;
                #pragma unroll
                for (int nb = 0; nb < 8; ++nb) {
                    Frag16 Bf;
                    #pragma unroll
                    for (int v = 0; v < 8; ++v) {
                        int k0 = kb * 32 + (v >> 2) * 16 + lh * 8 + (v & 3) * 2;
                        Bf.e[2*v]   = sW1[k0 * 128 + nb * 16 + lm];
                        Bf.e[2*v+1] = sW1[(k0 + 1) * 128 + nb * 16 + lm];
                    }
                    acc[nb].v = __builtin_amdgcn_wmma_f32_16x16x32_f16(
                        false, A.v, false, Bf.v, (short)0, acc[nb].v, false, false);
                }
            }
            #pragma unroll
            for (int nb = 0; nb < 8; ++nb)
                #pragma unroll
                for (int v = 0; v < 8; ++v)
                    sY1[wv][v + 8 * lh][nb * 16 + lm] =
                        (_Float16)fmaxf(acc[nb].f[v], 0.0f);
            __builtin_amdgcn_wave_barrier();

            // ---------------- GEMM2: [16 x 128] x [128 x 64] ---------------
            Acc8 a2[4];
            #pragma unroll
            for (int nb = 0; nb < 4; ++nb) {
                float bv = b2[nb * 16 + lm];
                #pragma unroll
                for (int v = 0; v < 8; ++v) a2[nb].f[v] = bv;
            }
            for (int kb = 0; kb < 4; ++kb) {
                Frag16 A;
                #pragma unroll
                for (int v = 0; v < 8; ++v) {
                    int k0 = kb * 32 + (v >> 2) * 16 + lh * 8 + (v & 3) * 2;
                    A.e[2*v]   = sY1[wv][lm][k0];
                    A.e[2*v+1] = sY1[wv][lm][k0 + 1];
                }
                const int kB = kb * 32 + lane;
                #pragma unroll
                for (int nb = 0; nb < 4; ++nb) {
                    Frag16 Bf;
                    #pragma unroll
                    for (int v = 0; v < 8; ++v) {
                        int k0 = kb * 32 + (v >> 2) * 16 + lh * 8 + (v & 3) * 2;
                        Bf.e[2*v]   = sW2[k0 * 64 + nb * 16 + lm];
                        Bf.e[2*v+1] = sW2[(k0 + 1) * 64 + nb * 16 + lm];
                    }
                    a2[nb].v = __builtin_amdgcn_wmma_f32_16x16x32_f16(
                        false, A.v, false, Bf.v, (short)0, a2[nb].v, false, false);
                }
            }
            #pragma unroll
            for (int nb = 0; nb < 4; ++nb)
                #pragma unroll
                for (int v = 0; v < 8; ++v)
                    sY0[wv][br][v + 8 * lh][nb * 16 + lm] = fmaxf(a2[nb].f[v], 0.0f);
            __builtin_amdgcn_wave_barrier();
        }

        // ---------------- heads: lanes 0-15, one sample each ---------------
        if (lh == 0) {
            const int m = lm, s = tile * 16 + m;
            if (s < N) {
                float l3a[3], l3b[3];
                #pragma unroll
                for (int j = 0; j < 3; ++j) { l3a[j] = pib[j]; l3b[j] = pib[j]; }
                float e = 2.0f * epb[0];
                for (int k = 0; k < 64; ++k) {
                    float ya = sY0[wv][0][m][k];   // y0   (normal)
                    float yb = sY0[wv][1][m][k];   // y0_  (mirror)
                    #pragma unroll
                    for (int j = 0; j < 3; ++j) {
                        l3a[j] += ya * piw[j * 64 + k];
                        l3b[j] += yb * piw[j * 64 + k];
                    }
                    e += (ya + yb) * epw[k];
                }
                float ma = fmaxf(l3a[0], fmaxf(l3a[1], l3a[2]));
                float ea0 = __expf(l3a[0]-ma), ea1 = __expf(l3a[1]-ma), ea2 = __expf(l3a[2]-ma);
                float sa = ea0 + ea1 + ea2;
                float pL0 = ea0/sa, pL1 = ea1/sa;               // open_long pre-flip
                float mb = fmaxf(l3b[0], fmaxf(l3b[1], l3b[2]));
                float eb0 = __expf(l3b[0]-mb), eb1 = __expf(l3b[1]-mb), eb2 = __expf(l3b[2]-mb);
                float sb = eb0 + eb1 + eb2;
                float pS0 = eb0/sb, pS1 = eb1/sb;               // open_short
                // open_long flipped = [p2,p1,p0]; [:,1:] = [p1,p0]
                float o0 = 0.5f * (pS0 + pL1);
                float o1 = 0.5f * (pS1 + pL0);
                float sh = -5.0f + amp[0] + 5.0f / (1.0f + __expf(-e));

                int c0 = idx[s*5], c1 = idx[s*5+1], c2 = idx[s*5+2],
                    c3 = idx[s*5+3], c4 = idx[s*5+4];
                bool condA = ((c1 == 5) || ((c1 == 7) && (c0 == 4)))
                             && (c2 == 11) && (c3 == 4);
                bool cond  = condA || ((c0 + c1 + c2 + c3 + c4) == 0);

                open_out[(long)s * 2]     = o0;
                open_out[(long)s * 2 + 1] = o1;
                ep_arr[s]  = sh;
                ind_arr[s] = cond ? 0.0f : 1.0f;
            }
        }
    }
}

// ---------------------------------------------------------------------------
// Kernel 2: sequential position scan. One block (128 thr / 4 waves) per batch
// element b; thread owns 5 circular-buffer slots in registers. One reduction
// (2 floats) per step: wave shfl-xor + parity-double-buffered LDS combine.
// ---------------------------------------------------------------------------
__global__ __launch_bounds__(128) void scan_kernel(
    const float* __restrict__ x2,       // rp = x2[:,0]
    const float* __restrict__ open0,    // d_out + N, (N,2)
    const float* __restrict__ ep_arr,
    const float* __restrict__ ind_arr,
    float* __restrict__ outraw,         // d_out[0..N)
    float* __restrict__ pmax,           // per-b max|out|
    int D, int H)
{
    const int b = blockIdx.x;
    const int t = threadIdx.x;
    const int lane = t & 31, wv = t >> 5;
    const int SL = 5;

    float pos0[SL], pos1[SL], pr0[SL], pr1[SL];
    #pragma unroll
    for (int j = 0; j < SL; ++j) { pos0[j] = pos1[j] = pr0[j] = pr1[j] = 0.0f; }

    __shared__ float red[2][4][2];
    int head = 0;
    float mx = 0.0f;

    for (int d = 0; d < D; ++d) {
        const long sidx = (long)b * D + d;
        const float rp  = x2[sidx * 8];
        const float ep  = ep_arr[sidx];
        const float op0 = open0[sidx * 2];
        const float op1 = open0[sidx * 2 + 1];
        const float ind = ind_arr[sidx];

        int newhead = head - 1; if (newhead < 0) newhead += H;

        float s0 = 0.0f, s1 = 0.0f;
        #pragma unroll
        for (int j = 0; j < SL; ++j) {
            int p = t * SL + j;
            if (p < H) {
                int age = p - head; if (age < 0) age += H;
                float ef = fminf(1.0f, 0.1f * (float)(H - 1 - age));
                float es = (1.0f - tanhf(fmaxf(fabsf(rp - pr0[j]) - pr1[j], 0.0f))) * ef;
                pos0[j] *= es; pos1[j] *= es;
                if (p != newhead) { s0 += pos0[j]; s1 += pos1[j]; }
            }
        }
        // wave allreduce
        #pragma unroll
        for (int off = 16; off > 0; off >>= 1) {
            s0 += __shfl_xor(s0, off, 32);
            s1 += __shfl_xor(s1, off, 32);
        }
        const int par = d & 1;
        if (lane == 0) { red[par][wv][0] = s0; red[par][wv][1] = s1; }
        __syncthreads();
        s0 = red[par][0][0] + red[par][1][0] + red[par][2][0] + red[par][3][0];
        s1 = red[par][0][1] + red[par][1][1] + red[par][2][1] + red[par][3][1];

        #pragma unroll
        for (int j = 0; j < SL; ++j) {
            int p = t * SL + j;
            if (p < H) {
                if (p == newhead) {
                    pos0[j] = (1.0f - s0) * op0;
                    pos1[j] = (1.0f - s1) * op1;
                    pr0[j] = rp; pr1[j] = ep;
                }
                pos0[j] *= ind; pos1[j] *= ind;
            }
        }
        if (t == 0) {
            float y0 = ind * (s0 + (1.0f - s0) * op0);
            float y1 = ind * (s1 + (1.0f - s1) * op1);
            float o  = y1 - y0;                     // -ch0 + ch1
            outraw[sidx] = o;
            mx = fmaxf(mx, fabsf(o));
        }
        head = newhead;
    }
    if (t == 0) pmax[b] = mx;
}

// ---------------------------------------------------------------------------
// Kernel 3: global max of per-b partials (re-reduced per block) + scale.
// ---------------------------------------------------------------------------
__global__ __launch_bounds__(256) void norm_kernel(
    float* __restrict__ out, const float* __restrict__ pmax, int B, int N)
{
    __shared__ float sm[256];
    const int t = threadIdx.x;
    float v = 0.0f;
    for (int i = t; i < B; i += 256) v = fmaxf(v, pmax[i]);
    sm[t] = v;
    __syncthreads();
    for (int off = 128; off > 0; off >>= 1) {
        if (t < off) sm[t] = fmaxf(sm[t], sm[t + off]);
        __syncthreads();
    }
    const float M = sm[0];
    const long i = (long)blockIdx.x * 256 + t;
    if (i < N) out[i] = out[i] / M;
}

// ---------------------------------------------------------------------------
extern "C" void kernel_launch(void* const* d_in, const int* in_sizes, int n_in,
                              void* d_out, int out_size, void* d_ws, size_t ws_size,
                              hipStream_t stream) {
    // setup_inputs() dict order: x0, x1m, x2, x3, idx, then params in insertion
    // order (conv00_w/b, conv10_w/b, fc11_w/b, fc12_w/b, pi_w/b, ep_w/b, amp,
    // oh0..oh4), then day_max.
    const float* x0  = (const float*)d_in[0];
    const float* x1m = (const float*)d_in[1];
    const float* x2  = (const float*)d_in[2];
    const float* x3  = (const float*)d_in[3];
    const int*   idx = (const int*)  d_in[4];
    const float* w00 = (const float*)d_in[5];
    const float* b00 = (const float*)d_in[6];
    const float* w10 = (const float*)d_in[7];
    const float* b10 = (const float*)d_in[8];
    const float* w1  = (const float*)d_in[9];
    const float* b1  = (const float*)d_in[10];
    const float* w2  = (const float*)d_in[11];
    const float* b2  = (const float*)d_in[12];
    const float* piw = (const float*)d_in[13];
    const float* pib = (const float*)d_in[14];
    const float* epw = (const float*)d_in[15];
    const float* epb = (const float*)d_in[16];
    const float* amp = (const float*)d_in[17];
    const float* oh0 = (const float*)d_in[18];
    const float* oh1 = (const float*)d_in[19];
    const float* oh2 = (const float*)d_in[20];
    const float* oh3 = (const float*)d_in[21];
    const float* oh4 = (const float*)d_in[22];

    const int N = in_sizes[2] / 8;          // x2 is (N, 8)
    const int D = 600;                       // day_max (reference constant)
    const int H = 600;                       // min(HOLD_MAX, D)
    const int B = N / D;

    float* out    = (float*)d_out;           // [0..N): pos_sum, [N..3N): open_0
    float* ws     = (float*)d_ws;
    float* ep_arr  = ws;
    float* ind_arr = ws + (size_t)N;
    float* pmaxbuf = ws + (size_t)2 * N;

    const int Ntiles = (N + 15) / 16;
    const int nblk1  = (Ntiles + WAVES - 1) / WAVES;

    mlp_kernel<<<nblk1, 128, 0, stream>>>(
        x0, x1m, x2, x3, idx, w00, b00, w10, b10, w1, b1, w2, b2,
        piw, pib, epw, epb, amp, oh0, oh1, oh2, oh3, oh4,
        out + (size_t)N, ep_arr, ind_arr, Ntiles, N);

    scan_kernel<<<B, 128, 0, stream>>>(
        x2, out + (size_t)N, ep_arr, ind_arr, out, pmaxbuf, D, H);

    norm_kernel<<<(N + 255) / 256, 256, 0, stream>>>(out, pmaxbuf, B, N);
}
